// TAUFusionMiddle_28887950033196
// MI455X (gfx1250) — compile-verified, matrix operand order checked by reading, not run on hardware
//
#include <hip/hip_runtime.h>
#include <hip/hip_bf16.h>

typedef __attribute__((ext_vector_type(16))) _Float16 v16h;
typedef __attribute__((ext_vector_type(8)))  float    v8f;
typedef uint32_t u32x4 __attribute__((ext_vector_type(4)));
typedef int      i32x4 __attribute__((ext_vector_type(4)));
typedef int      i32x8 __attribute__((ext_vector_type(8)));

#define D_MODEL 256
#define T_SEQ   8
#define N_HEADS 8
#define DK      32
#define DFF     1024
#define B_SZ    4
#define HW      2304     // 48*48
#define NSEQ    9216     // B*H*W
#define MROWS   73728    // NSEQ*T
#define NLAYERS 2

// LDS tile: 128 rows x 32 halfs, TDM-padded by 8 DWORDs per 16-DWORD row
// -> row stride 96B = 48 halfs (keeps 32B fragment reads aligned, spreads banks)
#define LDS_ROW 48

// ---------------------------------------------------------------------------
// TDM: DMA one 128x32 f16 tile (row stride = strideElems) from global to LDS.
// D# packing per CDNA5 ISA 8.3/8.4: group0 = {count|flags, lds_addr,
// global_addr[56:0], type=2}; group1 = {data_size=2B, pad 16DW->8DW,
// tensor/tile dims, dim0 stride}. Tracked on TENSORcnt.
// ---------------------------------------------------------------------------
__device__ inline void tdm_load_tile(uint32_t ldsOff, const _Float16* gptr,
                                     int strideElems) {
  uint64_t ga = (uint64_t)(uintptr_t)gptr;
  u32x4 g0;
  g0[0] = 1u;                                            // count=1, user mode
  g0[1] = ldsOff;                                        // lds_addr
  g0[2] = (uint32_t)ga;                                  // global_addr[31:0]
  g0[3] = (uint32_t)((ga >> 32) & 0x01FFFFFFu) | (2u << 30);  // addr[56:32], type=2
  i32x8 g1;
  g1[0] = (int)((1u << 16)      // data_size = 2 bytes
              | (1u << 20)      // pad_enable
              | (3u << 22)      // pad_interval = 16 DWORDs (one 64B row)
              | (7u << 25));    // pad_amount  = 8 DWORDs (32B)
  g1[1] = (int)(32u << 16);     // tensor_dim0[15:0] = 32 (bits 63:48)
  g1[2] = (int)(128u << 16);    // tensor_dim1[15:0] = 128 (bits 95:80)
  g1[3] = (int)(32u << 16);     // tile_dim0 = 32 (bits 127:112)
  g1[4] = (int)128u;            // tile_dim1 = 128, tile_dim2 = 0
  g1[5] = (int)strideElems;     // tensor_dim0_stride[31:0]
  g1[6] = 0;
  g1[7] = 0;
  i32x4 z4 = {0, 0, 0, 0};
#if defined(__clang_major__) && __clang_major__ >= 23
  i32x8 z8 = {0, 0, 0, 0, 0, 0, 0, 0};
  __builtin_amdgcn_tensor_load_to_lds(g0, g1, z4, z4, z8, 0);
#else
  __builtin_amdgcn_tensor_load_to_lds(g0, g1, z4, z4, 0);
#endif
}

// ---------------------------------------------------------------------------
// prep: (B,T,C,H,W) -> (n=b*HW+p, t, c) with LDS 32x32 transpose + sinusoidal PE
// writes fp32 residual stream h32 and f16 mirror h16
// ---------------------------------------------------------------------------
__global__ __launch_bounds__(256) void prep_kernel(const float* __restrict__ x,
                                                   float* __restrict__ h32,
                                                   _Float16* __restrict__ h16) {
  __shared__ float tile[32][33];
  const int bt = blockIdx.z;              // b*T + t
  const int b  = bt >> 3, t = bt & 7;
  const int pBase = blockIdx.x * 32;      // pixel tile
  const int cBase = blockIdx.y * 32;      // channel tile
  const int tx = threadIdx.x, ty = threadIdx.y;   // 32 x 8
#pragma unroll
  for (int i = 0; i < 4; ++i) {
    int c = cBase + ty + i * 8;
    tile[ty + i * 8][tx] = x[((size_t)bt * D_MODEL + c) * HW + pBase + tx];
  }
  __syncthreads();
  const float tf = (float)t;
#pragma unroll
  for (int i = 0; i < 4; ++i) {
    int p = pBase + ty + i * 8;
    int c = cBase + tx;
    // pe: div = exp(-(2i)*ln(10000)/d), even c -> sin, odd c -> cos
    float f  = __expf(-(float)(c & ~1) * (9.210340372f / 256.0f));
    float pe = (c & 1) ? __cosf(tf * f) : __sinf(tf * f);
    float val = tile[tx][ty + i * 8] + pe;
    size_t n   = (size_t)b * HW + p;
    size_t idx = (n * T_SEQ + t) * D_MODEL + c;
    h32[idx] = val;
    h16[idx] = (_Float16)val;
  }
}

// ---------------------------------------------------------------------------
// fp32 -> f16 weight conversion
// ---------------------------------------------------------------------------
__global__ void cvt_kernel(const float* __restrict__ s, _Float16* __restrict__ d, int n) {
  int i = blockIdx.x * 256 + threadIdx.x;
  if (i < n) d[i] = (_Float16)s[i];
}

// ---------------------------------------------------------------------------
// WMMA GEMM with TDM double-buffered LDS staging:
//   out = A(MxK,f16) * W(NxK,f16)^T + bias
//   block tile 128x128; wave 0 DMAs the 128x32 A and B tiles for stage i+1
//   via tensor_load_to_lds while all 8 waves run WMMA on stage i from LDS.
//   EPI 0: store f16        (QKV projections)
//   EPI 1: GELU -> f16      (FFN up-projection)
//   EPI 2: +residual -> f32 (attention out / FFN down, pre-LayerNorm)
// ---------------------------------------------------------------------------
template <int EPI>
__global__ __launch_bounds__(256) void gemm_kernel(const _Float16* __restrict__ A,
                                                   const _Float16* __restrict__ W,
                                                   const float* __restrict__ bias,
                                                   const float* __restrict__ res,
                                                   _Float16* __restrict__ out16,
                                                   float* __restrict__ out32,
                                                   int M, int N, int K) {
  __shared__ _Float16 smem[2][2][128 * LDS_ROW];  // [stage][A/B][tile]
  const int wave = threadIdx.x >> 5;   // 0..7
  const int lane = threadIdx.x & 31;
  const int wm = wave >> 1;            // 0..3 (M)
  const int wn = wave & 1;             // 0..1 (N)
  const int lm = lane & 15;
  const int lk = (lane >> 4) * 16;     // K half selected by lane group
  const int nk = K >> 5;

  const _Float16* Abase = A + (size_t)blockIdx.x * 128 * K;
  const _Float16* Wbase = W + (size_t)blockIdx.y * 128 * K;

  if (wave == 0) {
    tdm_load_tile((uint32_t)(uintptr_t)&smem[0][0][0], Abase, K);
    tdm_load_tile((uint32_t)(uintptr_t)&smem[0][1][0], Wbase, K);
  }

  v8f c[2][4] = {};
  for (int i = 0; i < nk; ++i) {
    if (wave == 0) {
      if (i + 1 < nk) {
        const int kb = (i + 1) << 5;
        tdm_load_tile((uint32_t)(uintptr_t)&smem[(i + 1) & 1][0][0], Abase + kb, K);
        tdm_load_tile((uint32_t)(uintptr_t)&smem[(i + 1) & 1][1][0], Wbase + kb, K);
        __builtin_amdgcn_s_wait_tensorcnt(2);  // stage i landed; i+1 in flight
      } else {
        __builtin_amdgcn_s_wait_tensorcnt(0);
      }
    }
    __syncthreads();  // release all waves once stage i is in LDS

    const _Float16* As = &smem[i & 1][0][0];
    const _Float16* Bs = &smem[i & 1][1][0];
    v16h a[2], bfr[4];
#pragma unroll
    for (int ii = 0; ii < 2; ++ii)
      a[ii] = *(const v16h*)(As + (wm * 32 + ii * 16 + lm) * LDS_ROW + lk);
#pragma unroll
    for (int j = 0; j < 4; ++j)
      bfr[j] = *(const v16h*)(Bs + (wn * 64 + j * 16 + lm) * LDS_ROW + lk);
#pragma unroll
    for (int ii = 0; ii < 2; ++ii)
#pragma unroll
      for (int j = 0; j < 4; ++j)
        c[ii][j] = __builtin_amdgcn_wmma_f32_16x16x32_f16(
            false, a[ii], false, bfr[j], (short)0, c[ii][j], false, false);

    __syncthreads();  // all reads of stage i done before its buffer is reused
  }

  const int rowOff = (lane >> 4) * 8;  // lanes 16-31 hold M = 8..15 of each tile
#pragma unroll
  for (int i = 0; i < 2; ++i) {
#pragma unroll
    for (int j = 0; j < 4; ++j) {
      const int col = blockIdx.y * 128 + wn * 64 + j * 16 + lm;
      const float bv = bias[col];
#pragma unroll
      for (int r = 0; r < 8; ++r) {
        const int row = blockIdx.x * 128 + wm * 32 + i * 16 + rowOff + r;
        float v = c[i][j][r] + bv;
        const size_t idx = (size_t)row * N + col;
        if (EPI == 0) {
          out16[idx] = (_Float16)v;
        } else if (EPI == 1) {
          v = 0.5f * v * (1.0f + erff(v * 0.70710678118f));  // exact erf GELU
          out16[idx] = (_Float16)v;
        } else {
          out32[idx] = v + res[idx];
        }
      }
    }
  }
}

// ---------------------------------------------------------------------------
// causal MHSA for T=8, dk=32; one thread per (seq, head, t)
// ---------------------------------------------------------------------------
__global__ __launch_bounds__(256) void attn_kernel(const _Float16* __restrict__ q,
                                                   const _Float16* __restrict__ k,
                                                   const _Float16* __restrict__ v,
                                                   _Float16* __restrict__ ctx) {
  int tid = blockIdx.x * 256 + threadIdx.x;
  const int t = tid & 7;
  const int hh = (tid >> 3) & 7;
  const int n = tid >> 6;
  const _Float16* qp = q + ((size_t)n * T_SEQ + t) * D_MODEL + hh * DK;
  float qv[DK];
#pragma unroll
  for (int j = 0; j < DK; ++j) qv[j] = (float)qp[j];
  float sc[T_SEQ];
  float mx = -1e30f;
  for (int s = 0; s <= t; ++s) {
    const _Float16* kp = k + ((size_t)n * T_SEQ + s) * D_MODEL + hh * DK;
    float acc = 0.0f;
#pragma unroll
    for (int j = 0; j < DK; ++j) acc += qv[j] * (float)kp[j];
    acc *= 0.1767766953f;  // 1/sqrt(32)
    sc[s] = acc;
    mx = fmaxf(mx, acc);
  }
  float sum = 0.0f;
  for (int s = 0; s <= t; ++s) { sc[s] = __expf(sc[s] - mx); sum += sc[s]; }
  const float inv = 1.0f / sum;
  float cx[DK];
#pragma unroll
  for (int j = 0; j < DK; ++j) cx[j] = 0.0f;
  for (int s = 0; s <= t; ++s) {
    const _Float16* vp = v + ((size_t)n * T_SEQ + s) * D_MODEL + hh * DK;
    const float w = sc[s] * inv;
#pragma unroll
    for (int j = 0; j < DK; ++j) cx[j] += w * (float)vp[j];
  }
  _Float16* cp = ctx + ((size_t)n * T_SEQ + t) * D_MODEL + hh * DK;
#pragma unroll
  for (int j = 0; j < DK; ++j) cp[j] = (_Float16)cx[j];
}

// ---------------------------------------------------------------------------
// LayerNorm over d=256: one wave per row (wave32, 8 elems/lane, shfl_xor tree)
// writes fp32 residual stream + f16 mirror
// ---------------------------------------------------------------------------
__global__ __launch_bounds__(256) void ln_kernel(const float* __restrict__ in,
                                                 const float* __restrict__ g,
                                                 const float* __restrict__ be,
                                                 float* __restrict__ out32,
                                                 _Float16* __restrict__ out16) {
  const int wave = threadIdx.x >> 5, lane = threadIdx.x & 31;
  const int row = blockIdx.x * 8 + wave;
  const float* p = in + (size_t)row * D_MODEL + lane * 8;
  float x[8];
#pragma unroll
  for (int i = 0; i < 8; ++i) x[i] = p[i];
  float s = 0.0f, s2 = 0.0f;
#pragma unroll
  for (int i = 0; i < 8; ++i) { s += x[i]; s2 += x[i] * x[i]; }
#pragma unroll
  for (int off = 16; off; off >>= 1) {
    s  += __shfl_xor(s, off, 32);
    s2 += __shfl_xor(s2, off, 32);
  }
  const float mean = s * (1.0f / 256.0f);
  const float var  = s2 * (1.0f / 256.0f) - mean * mean;  // biased var
  const float rstd = rsqrtf(var + 1e-5f);
  const int cb = lane * 8;
#pragma unroll
  for (int i = 0; i < 8; ++i) {
    float y = (x[i] - mean) * rstd * g[cb + i] + be[cb + i];
    size_t idx = (size_t)row * D_MODEL + cb + i;
    out32[idx] = y;
    out16[idx] = (_Float16)y;
  }
}

// ---------------------------------------------------------------------------
// take last timestep, back to (B, C, H, W)
// ---------------------------------------------------------------------------
__global__ void final_kernel(const float* __restrict__ h32, float* __restrict__ out) {
  int tid = blockIdx.x * 256 + threadIdx.x;
  int ww = tid % 48;
  int r = tid / 48;
  int hhp = r % 48; r /= 48;
  int c = r % D_MODEL;
  int b = r / D_MODEL;
  size_t n = ((size_t)b * 48 + hhp) * 48 + ww;
  out[tid] = h32[(n * T_SEQ + (T_SEQ - 1)) * D_MODEL + c];
}

// ---------------------------------------------------------------------------
extern "C" void kernel_launch(void* const* d_in, const int* in_sizes, int n_in,
                              void* d_out, int out_size, void* d_ws, size_t ws_size,
                              hipStream_t stream) {
  const float* x  = (const float*)d_in[0];
  const float* Wq = (const float*)d_in[1];
  const float* bq = (const float*)d_in[2];
  const float* Wk = (const float*)d_in[3];
  const float* bk = (const float*)d_in[4];
  const float* Wv = (const float*)d_in[5];
  const float* bv = (const float*)d_in[6];
  const float* Wo = (const float*)d_in[7];
  const float* bo = (const float*)d_in[8];
  const float* g1 = (const float*)d_in[9];
  const float* be1 = (const float*)d_in[10];
  const float* W1 = (const float*)d_in[11];
  const float* b1 = (const float*)d_in[12];
  const float* W2 = (const float*)d_in[13];
  const float* b2 = (const float*)d_in[14];
  const float* g2 = (const float*)d_in[15];
  const float* be2 = (const float*)d_in[16];

  char* ws = (char*)d_ws;
  size_t off = 0;
  auto take = [&](size_t bytes) -> char* {
    char* p = ws + off;
    off += (bytes + 255) & ~(size_t)255;
    return p;
  };
  float*    h32   = (float*)take((size_t)MROWS * D_MODEL * 4);
  _Float16* h16   = (_Float16*)take((size_t)MROWS * D_MODEL * 2);
  _Float16* q16   = (_Float16*)take((size_t)MROWS * D_MODEL * 2);
  _Float16* k16   = (_Float16*)take((size_t)MROWS * D_MODEL * 2);
  _Float16* v16   = (_Float16*)take((size_t)MROWS * D_MODEL * 2);
  _Float16* ctx16 = (_Float16*)take((size_t)MROWS * D_MODEL * 2);
  float*    tmp32 = (float*)take((size_t)MROWS * D_MODEL * 4);
  _Float16* ff1   = (_Float16*)take((size_t)MROWS * DFF * 2);
  const int WDD = NLAYERS * D_MODEL * D_MODEL;     // 131072
  const int WFF = NLAYERS * DFF * D_MODEL;         // 524288
  _Float16* wq16 = (_Float16*)take((size_t)WDD * 2);
  _Float16* wk16 = (_Float16*)take((size_t)WDD * 2);
  _Float16* wv16 = (_Float16*)take((size_t)WDD * 2);
  _Float16* wo16 = (_Float16*)take((size_t)WDD * 2);
  _Float16* w116 = (_Float16*)take((size_t)WFF * 2);
  _Float16* w216 = (_Float16*)take((size_t)WFF * 2);

  // weights -> f16
  cvt_kernel<<<(WDD + 255) / 256, 256, 0, stream>>>(Wq, wq16, WDD);
  cvt_kernel<<<(WDD + 255) / 256, 256, 0, stream>>>(Wk, wk16, WDD);
  cvt_kernel<<<(WDD + 255) / 256, 256, 0, stream>>>(Wv, wv16, WDD);
  cvt_kernel<<<(WDD + 255) / 256, 256, 0, stream>>>(Wo, wo16, WDD);
  cvt_kernel<<<(WFF + 255) / 256, 256, 0, stream>>>(W1, w116, WFF);
  cvt_kernel<<<(WFF + 255) / 256, 256, 0, stream>>>(W2, w216, WFF);

  // transpose + positional encoding
  prep_kernel<<<dim3(HW / 32, D_MODEL / 32, B_SZ * T_SEQ), dim3(32, 8), 0, stream>>>(x, h32, h16);

  const dim3 gDD(MROWS / 128, D_MODEL / 128);  // (576, 2)
  const dim3 gDF(MROWS / 128, DFF / 128);      // (576, 8)

  for (int l = 0; l < NLAYERS; ++l) {
    const int oDD = l * D_MODEL * D_MODEL;
    const int oFF = l * DFF * D_MODEL;
    // QKV projections
    gemm_kernel<0><<<gDD, 256, 0, stream>>>(h16, wq16 + oDD, bq + l * D_MODEL, nullptr,
                                            q16, nullptr, MROWS, D_MODEL, D_MODEL);
    gemm_kernel<0><<<gDD, 256, 0, stream>>>(h16, wk16 + oDD, bk + l * D_MODEL, nullptr,
                                            k16, nullptr, MROWS, D_MODEL, D_MODEL);
    gemm_kernel<0><<<gDD, 256, 0, stream>>>(h16, wv16 + oDD, bv + l * D_MODEL, nullptr,
                                            v16, nullptr, MROWS, D_MODEL, D_MODEL);
    // causal attention
    attn_kernel<<<(NSEQ * N_HEADS * T_SEQ) / 256, 256, 0, stream>>>(q16, k16, v16, ctx16);
    // Wo projection + residual
    gemm_kernel<2><<<gDD, 256, 0, stream>>>(ctx16, wo16 + oDD, bo + l * D_MODEL, h32,
                                            nullptr, tmp32, MROWS, D_MODEL, D_MODEL);
    ln_kernel<<<MROWS / 8, 256, 0, stream>>>(tmp32, g1 + l * D_MODEL, be1 + l * D_MODEL, h32, h16);
    // FFN
    gemm_kernel<1><<<gDF, 256, 0, stream>>>(h16, w116 + oFF, b1 + l * DFF, nullptr,
                                            ff1, nullptr, MROWS, DFF, D_MODEL);
    gemm_kernel<2><<<gDD, 256, 0, stream>>>(ff1, w216 + oFF, b2 + l * D_MODEL, h32,
                                            nullptr, tmp32, MROWS, D_MODEL, DFF);
    ln_kernel<<<MROWS / 8, 256, 0, stream>>>(tmp32, g2 + l * D_MODEL, be2 + l * D_MODEL, h32, h16);
  }

  final_kernel<<<(B_SZ * D_MODEL * HW) / 256, 256, 0, stream>>>(h32, (float*)d_out);
}
